// KAN_Convolution_11321533792346
// MI455X (gfx1250) — matrix operand order repaired and checked by reading, not compiled, not run on hardware
//
#include <hip/hip_runtime.h>
#include <hip/hip_bf16.h>

typedef __attribute__((ext_vector_type(16))) _Float16 v16h;
typedef __attribute__((ext_vector_type(8)))  _Float16 v8h;
typedef __attribute__((ext_vector_type(8)))  float    v8f;

#define TILE 16
#define ROWH 104   // halves per A-row: 208 bytes -> 16B aligned, bank-conflict-free b128 reads

__global__ __launch_bounds__(256)
void kan_conv_wmma_kernel(const float* __restrict__ x,
                          const float* __restrict__ base_w,
                          const float* __restrict__ spline_w,
                          const float* __restrict__ spline_s,
                          float* __restrict__ out)
{
    // 256 A-rows (one per thread) + fused f16 weight vector (K padded 81 -> 96)
    __shared__ __align__(16) _Float16 sA[256 * ROWH];   // 53248 B
    __shared__ __align__(32) _Float16 sW[96];

    const int t    = threadIdx.x;
    const int wave = t >> 5;
    const int lane = t & 31;

    // ---- fused weights: [0,72) = spline_weight*spline_scaler, [72,81) = base_weight, rest 0
    if (t < 96) {
        _Float16 w = (_Float16)0.f;
        if (t < 72)      w = (_Float16)(spline_w[t] * spline_s[t >> 3]);
        else if (t < 81) w = (_Float16)base_w[t - 72];
        sW[t] = w;
    }
    __syncthreads();

    const int ox0   = blockIdx.x * TILE;
    const int oy0   = blockIdx.y * TILE;
    const int plane = blockIdx.z;                       // b*16 + c
    const float* xp = x + (size_t)plane * 96 * 96;

    // thread -> output pixel: set 0 = lanes 0-15 (row = ox), set 1 = lanes 16-31
    const int oy = oy0 + 2 * wave + (lane >> 4);
    const int ox = ox0 + (lane & 15);

    // ---- zero my A-row (12 x b128) ----
    _Float16* row = &sA[(wave * 32 + lane) * ROWH];
    {
        uint4 z; z.x = z.y = z.z = z.w = 0u;
        uint4* rp = (uint4*)row;
        #pragma unroll
        for (int q = 0; q < 12; ++q) rp[q] = z;
    }

    // ---- elementwise phase: 9 taps, closed-form uniform cubic B-spline (4 nonzero bases) ----
    #pragma unroll
    for (int ky = 0; ky < 3; ++ky) {
        #pragma unroll
        for (int kx = 0; kx < 3; ++kx) {
            const int j  = ky * 3 + kx;
            const int iy = min(oy + ky, 95);            // clamp: edge threads compute garbage rows
            const int ix = min(ox + kx, 95);            // that are never stored
            const float v = xp[iy * 96 + ix];

            row[72 + j] = (_Float16)(v / (1.f + __expf(-v)));   // SiLU slot

            const float tt = (v + 2.2f) * 2.5f;         // (v - g0) / h, g0=-2.2, h=0.4
            const float cf = floorf(tt);
            const int   ci = (int)cf;
            const float u  = tt - cf;
            const float u2 = u * u, u3 = u2 * u;
            const float omu = 1.f - u;
            const float B0 = omu * omu * omu * (1.f / 6.f);
            const float B1 = 0.5f * u3 - u2 + (2.f / 3.f);
            const float B2 = 0.5f * (u + u2 - u3) + (1.f / 6.f);
            const float B3 = u3 * (1.f / 6.f);
            if (ci >= 0 && ci <= 10) {                  // inside extended grid [-2.2, 2.2)
                const int i0 = ci - 3;
                _Float16* bp = row + 8 * j;
                if ((unsigned)(i0 + 0) < 8u) bp[i0 + 0] = (_Float16)B0;
                if ((unsigned)(i0 + 1) < 8u) bp[i0 + 1] = (_Float16)B1;
                if ((unsigned)(i0 + 2) < 8u) bp[i0 + 2] = (_Float16)B2;
                if ((unsigned)(i0 + 3) < 8u) bp[i0 + 3] = (_Float16)B3;
            }
        }
    }
    // A-rows are wave-private; same-wave LDS ops are in-order. Fence the compiler only.
    __builtin_amdgcn_wave_barrier();

    // ---- B fragments: weight vector replicated across all 16 N-columns ----
    // 16-bit B 32x16: VGPR v holds K = 2v,2v+1 (lanes 0-15) / K = 2v+16,2v+17 (lanes 16-31)
    const int khalf = (lane < 16) ? 0 : 16;
    const v16h bf0 = *(const v16h*)(&sW[ 0 + khalf]);
    const v16h bf1 = *(const v16h*)(&sW[32 + khalf]);
    const v16h bf2 = *(const v16h*)(&sW[64 + khalf]);

    // 16-bit A 16x32: lane L -> row L%16; VGPR0-3 = K 0-7 (lo half) / 16-23 (hi half),
    // VGPR4-7 = K 8-15 / 24-31 (contiguous, matching B)
    const int aoff1 = (lane < 16) ? 0 : 16;
    const int aoff2 = (lane < 16) ? 8 : 24;

    #pragma unroll
    for (int s = 0; s < 2; ++s) {
        const _Float16* arow = &sA[(wave * 32 + s * 16 + (lane & 15)) * ROWH];
        v8f acc = {};
        #pragma unroll
        for (int ch = 0; ch < 3; ++ch) {
            const v8h lo = *(const v8h*)(arow + ch * 32 + aoff1);
            const v8h hi = *(const v8h*)(arow + ch * 32 + aoff2);
            const v16h a = __builtin_shufflevector(lo, hi,
                0, 1, 2, 3, 4, 5, 6, 7, 8, 9, 10, 11, 12, 13, 14, 15);
            const v16h b = (ch == 0) ? bf0 : (ch == 1) ? bf1 : bf2;
            acc = __builtin_amdgcn_wmma_f32_16x16x32_f16(
                false, a, false, b, (short)0, acc, false, false);
        }
        // D layout: lanes 0-15 elem r = row r; lanes 16-31 elem r = row r+8.
        // All columns identical (B replicated), so any lane in a half has its 8 rows.
        const int e = lane & 7;
        float val = acc[0];
        val = (e == 1) ? acc[1] : val;
        val = (e == 2) ? acc[2] : val;
        val = (e == 3) ? acc[3] : val;
        val = (e == 4) ? acc[4] : val;
        val = (e == 5) ? acc[5] : val;
        val = (e == 6) ? acc[6] : val;
        val = (e == 7) ? acc[7] : val;
        const bool active = (lane < 8) || (lane >= 16 && lane < 24);
        const int  r   = (lane < 8) ? lane : lane - 8;   // row index == ox offset
        const int  oys = oy0 + 2 * wave + s;
        const int  oxs = ox0 + r;
        if (active && oys < 94 && oxs < 94)
            out[(size_t)plane * (94 * 94) + oys * 94 + oxs] = val;
    }
}

extern "C" void kernel_launch(void* const* d_in, const int* in_sizes, int n_in,
                              void* d_out, int out_size, void* d_ws, size_t ws_size,
                              hipStream_t stream) {
    const float* x        = (const float*)d_in[0];
    // d_in[1] = grid: uniform, constants (g0=-2.2, h=0.4) baked into the kernel
    const float* base_w   = (const float*)d_in[2];
    const float* spline_w = (const float*)d_in[3];
    const float* spline_s = (const float*)d_in[4];
    float* out            = (float*)d_out;

    dim3 grid(6, 6, 128);   // ceil(94/16)^2 tiles x (B*C) planes
    dim3 block(256);        // 8 waves; each wave: 2 WMMA sets of 16 outputs
    kan_conv_wmma_kernel<<<grid, block, 0, stream>>>(x, base_w, spline_w, spline_s, out);
}
